// F0Extractor_73091753443819
// MI455X (gfx1250) — matrix operand order repaired and checked by reading, not compiled, not run on hardware
//
#include <hip/hip_runtime.h>

typedef __attribute__((ext_vector_type(16))) _Float16 v16h;
typedef __attribute__((ext_vector_type(8)))  float    v8f;

#define SRATE      16000.0f
#define HOPSZ      256
#define FRAME_LEN  512
#define NFRAMES    641
#define T_LEN      163840
#define PADSZ      256
#define LDS_FRONT  240   // zeros so B[k][u]=x[k-16u] can read 16*15 below the frame
#define LDS_TOTAL  768   // 240 front zeros + 512 samples + 16 tail zeros
#define WPB        8     // waves (frames) per block

__global__ __launch_bounds__(256)
void f0_autocorr_wmma(const float* __restrict__ x, float* __restrict__ out) {
    __shared__ _Float16 lds[WPB][LDS_TOTAL];

    const int lane     = threadIdx.x & 31;
    const int wave     = threadIdx.x >> 5;
    const int frame_id = blockIdx.x * WPB + wave;          // grid sized exactly: 41024 frames
    const int b        = frame_id / NFRAMES;
    const int f        = frame_id - b * NFRAMES;
    const float* __restrict__ xb = x + (long)b * T_LEN;
    _Float16* xl = lds[wave];

    // ---- zero the pad regions of this wave's LDS strip ----
    for (int t = lane; t < LDS_FRONT; t += 32) xl[t] = (_Float16)0.0f;
    if (lane < 16) xl[LDS_FRONT + FRAME_LEN + lane] = (_Float16)0.0f;

    // ---- stage frame (f32 -> f16) into LDS; reflection only on edge frames ----
    const int base = f * HOPSZ - PADSZ;                    // signal index of frame sample 0
    if (f > 0 && f < NFRAMES - 1) {
        // interior frame: fully in-range, 64B-aligned vector loads
        const float4* src = (const float4*)(xb + base + lane * 16);
        #pragma unroll
        for (int i = 0; i < 4; ++i) {
            float4 vv = src[i];
            int o = LDS_FRONT + lane * 16 + i * 4;
            xl[o + 0] = (_Float16)vv.x; xl[o + 1] = (_Float16)vv.y;
            xl[o + 2] = (_Float16)vv.z; xl[o + 3] = (_Float16)vv.w;
        }
    } else {
        #pragma unroll
        for (int i = 0; i < 16; ++i) {
            int j = lane * 16 + i;
            int g = base + j;
            g = (g < 0) ? -g : g;                          // reflect low (no edge repeat)
            g = (g >= T_LEN) ? (2 * T_LEN - 2 - g) : g;    // reflect high
            xl[LDS_FRONT + j] = (_Float16)xb[g];
        }
    }
    __syncthreads();

    // ---- accumulate D[v][u] = sum_k x[k+v] * x[k-16u]  ==  ac[16u+v] ----
    const int vrow = lane & 15;     // A row (Hankel shift v) == B column u for this lane
    const int h    = lane >> 4;     // lane half-group
    v8f acc = {};

    #pragma unroll
    for (int c = 0; c < 16; ++c) {
        const int k0 = c * 32;

        // A fragment: 16-bit A 16x32 layout.
        // element e in [0,8):  K = h*8 + e        -> x[k0 + h*8 + e + v]
        // element 8+e:         K = 16 + h*8 + e   -> x[k0 + 16 + h*8 + e + v]
        v16h a;
        {
            const int s = LDS_FRONT + k0 + vrow + h * 8;
            #pragma unroll
            for (int e = 0; e < 8; ++e) {
                a[e]     = xl[s + e];
                a[8 + e] = xl[s + 16 + e];
            }
        }

        // B fragment: same K order as A; element e<8 -> K = 8h + e, element 8+e -> K = 16 + 8h + e.
        // -> x[k0 + K - 16u]; start is 8-half aligned.
        v16h bf;
        {
            const int s = LDS_FRONT + k0 + h * 8 - vrow * 16;    // >= 0, multiple of 8
            const uint4* p = (const uint4*)(xl + s);             // 16B-aligned
            union { uint4 q[2]; v16h hv; } ub;
            ub.q[0] = p[0];
            ub.q[1] = p[2];
            bf = ub.hv;
        }

        acc = __builtin_amdgcn_wmma_f32_16x16x32_f16(
            /*neg_a=*/false, a, /*neg_b=*/false, bf,
            /*c_mod=*/(short)0, acc, /*reuse_a=*/false, /*reuse_b=*/false);
    }

    // ---- argmax over lags p = 16u + v', u = lane&15 (column), v' = 8h + r (row) ----
    float best  = -3.4e38f;
    int   bestP = 1 << 30;
    const int u = lane & 15;
    if (u >= 2) {                                  // lags < 32 excluded
        #pragma unroll
        for (int r = 0; r < 8; ++r) {
            int   p   = 16 * u + 8 * h + r;
            float val = acc[r];
            if (val > best || (val == best && p < bestP)) { best = val; bestP = p; }
        }
    }
    #pragma unroll
    for (int m = 16; m >= 1; m >>= 1) {
        float ov = __shfl_xor(best, m, 32);
        int   op = __shfl_xor(bestP, m, 32);
        if (ov > best || (ov == best && op < bestP)) { best = ov; bestP = op; }
    }

    if (lane == 0) {
        float f0 = SRATE / ((float)bestP + 1e-8f);
        f0 = fminf(fmaxf(f0, 50.0f), 500.0f);
        out[frame_id] = f0;
    }
}

extern "C" void kernel_launch(void* const* d_in, const int* in_sizes, int n_in,
                              void* d_out, int out_size, void* d_ws, size_t ws_size,
                              hipStream_t stream) {
    (void)in_sizes; (void)n_in; (void)d_ws; (void)ws_size; (void)out_size;
    const float* x  = (const float*)d_in[0];
    float*       out = (float*)d_out;
    const int total_frames = 64 * NFRAMES;        // 41024
    dim3 grid(total_frames / WPB);                // 5128 blocks, exact
    f0_autocorr_wmma<<<grid, 256, 0, stream>>>(x, out);
}
